// TransICD_29128468201753
// MI455X (gfx1250) — matrix operand order repaired and checked by reading, not compiled, not run on hardware
//
#include <hip/hip_runtime.h>
#include <cstdint>

typedef __attribute__((ext_vector_type(16))) __bf16 v16bf;
typedef __attribute__((ext_vector_type(8)))  float  v8f;
typedef __attribute__((ext_vector_type(4))) unsigned tdm_g0_t;
typedef __attribute__((ext_vector_type(8))) int      tdm_g1_t;
typedef __attribute__((ext_vector_type(4))) int      tdm_g23_t;

#if defined(__HIP_DEVICE_COMPILE__) && __has_builtin(__builtin_amdgcn_tensor_load_to_lds)
#define TICD_TDM 1
#else
#define TICD_TDM 0
#endif

namespace ticd {

constexpr int B = 8, S = 1024, E = 512, H = 8, L = 2, Cc = 8921, DFF = 2048, DH = 64;
constexpr int LD3E = 3 * E;   // qkv row stride
constexpr int RSTR = 34;      // LDS B-tile row stride in elements (68B: 17 banks, conflict-free)

__device__ inline __bf16 f2bf(float f) {
  unsigned u = __builtin_bit_cast(unsigned, f);
  unsigned r = (u + 0x7FFFu + ((u >> 16) & 1u)) >> 16;
  unsigned short us = (unsigned short)r;
  return __builtin_bit_cast(__bf16, us);
}

__device__ inline v8f vzero8() {
  v8f z;
#pragma unroll
  for (int i = 0; i < 8; ++i) z[i] = 0.f;
  return z;
}

#if TICD_TDM
// Issue a TDM load of a [rows x 32] bf16 tile (row stride ldb elements) into LDS.
// LDS gets 1 DWORD pad per 16 DWORDs -> row stride RSTR=34 elements.
// Rows >= rows_total read as zero (OOB).  D# packing per CDNA5 ISA ch.8.
__device__ inline void tdm_load_b_tile(const __bf16* gsrc, unsigned lds_off,
                                       int rows_total, int ldb) {
  unsigned long long ga = (unsigned long long)(uintptr_t)gsrc;
  tdm_g0_t g0;
  g0[0] = 1u;                                   // count=1, user descriptor
  g0[1] = lds_off;                              // lds_addr (bytes)
  g0[2] = (unsigned)ga;                         // global_addr[31:0]
  g0[3] = (unsigned)((ga >> 32) & 0x1FFFFFFull) | (2u << 30);  // addr[56:32] | type=2
  unsigned rt = rows_total < 0 ? 0u : (unsigned)rows_total;
  tdm_g1_t g1;
  g1[0] = (int)0x00D10000u;                     // data_size=2B, pad_enable, pad_interval=16 DW
  g1[1] = (int)(32u << 16);                     // tensor_dim0 = 32 (lo16)
  g1[2] = (int)((rt & 0xFFFFu) << 16);          // tensor_dim0 hi=0 | tensor_dim1 lo16
  g1[3] = (int)(((rt >> 16) & 0xFFFFu) | (32u << 16));  // tensor_dim1 hi | tile_dim0=32
  g1[4] = (int)256u;                            // tile_dim1=256, tile_dim2=0
  g1[5] = (int)(unsigned)ldb;                   // tensor_dim0_stride lo32
  g1[6] = 0;                                    // stride hi | dim1_stride lo (unused, 2D)
  g1[7] = 0;
  tdm_g23_t gz;
  gz[0] = 0; gz[1] = 0; gz[2] = 0; gz[3] = 0;
#if __clang_major__ >= 23
  tdm_g1_t gz8;
#pragma unroll
  for (int i = 0; i < 8; ++i) gz8[i] = 0;
  __builtin_amdgcn_tensor_load_to_lds(g0, g1, gz, gz, gz8, 0);
#else
  __builtin_amdgcn_tensor_load_to_lds(g0, g1, gz, gz, 0);
#endif
}
#endif

// ------------------------------------------------------------------
// f32 -> bf16 convert (weights)
// ------------------------------------------------------------------
__global__ void cvt_kernel(const float* __restrict__ in, __bf16* __restrict__ out, long n) {
  long i = (long)blockIdx.x * blockDim.x + threadIdx.x;
  long stride = (long)gridDim.x * blockDim.x;
  for (; i < n; i += stride) out[i] = f2bf(in[i]);
}

// ------------------------------------------------------------------
// embed: x = inputs*sqrt(E) + pe[b]  (pe indexed by batch: source bug kept)
// ------------------------------------------------------------------
__global__ void embed_kernel(const float* __restrict__ inp, const float* __restrict__ pe,
                             float* __restrict__ x, __bf16* __restrict__ xb) {
  long i = (long)blockIdx.x * blockDim.x + threadIdx.x;
  const long n = (long)B * S * E;
  if (i >= n) return;
  int b = (int)(i / ((long)S * E));
  int e = (int)(i % E);
  float v = inp[i] * 22.62741699796952f + pe[b * E + e];
  x[i] = v;
  xb[i] = f2bf(v);
}

// ------------------------------------------------------------------
// vT[(b*H+h)*DH + d][s] = V[b,s,h,d]   (from qkv section 2)
// ------------------------------------------------------------------
__global__ void transpose_v_kernel(const __bf16* __restrict__ qkv, __bf16* __restrict__ vT) {
  long i = (long)blockIdx.x * blockDim.x + threadIdx.x;
  const long n = (long)B * E * S;
  if (i >= n) return;
  int s = (int)(i % S);
  long rest = i / S;
  int d = (int)(rest % DH);
  int h = (int)((rest / DH) % H);
  int b = (int)(rest / ((long)DH * H));
  vT[i] = qkv[((long)b * S + s) * LD3E + 2 * E + h * DH + d];
}

// ------------------------------------------------------------------
// xbT[b][e][s] = xb[b][s][e]   (for the attn@x GEMM)
// ------------------------------------------------------------------
__global__ void transpose_x_kernel(const __bf16* __restrict__ xb, __bf16* __restrict__ xbT) {
  long i = (long)blockIdx.x * blockDim.x + threadIdx.x;
  const long n = (long)B * E * S;
  if (i >= n) return;
  int s = (int)(i % S);
  int e = (int)((i / S) % E);
  int b = (int)(i / ((long)S * E));
  xbT[i] = xb[((long)b * S + s) * E + e];
}

// ------------------------------------------------------------------
// WMMA GEMM:  C[M,N] = act( A[M,K] * B^T + bias ),  B weight-layout [N,K].
//   AF32   : A is fp32, converted to bf16 on load
//   STORET : store transposed-batched: row m -> (b=m/Sdim, s=m%Sdim),
//            Cf[(b*N + n)*Sdim + s]   (label scores -> d_out [B,C,S])
// Block: 256 threads = 8 waves (2 M x 4 N); wave tile 32x64; block tile 64x256.
// B tile (256x32) staged into LDS by the Tensor Data Mover, double buffered,
// tracked with TENSORcnt; fragments read from LDS conflict-free (RSTR=34).
// ------------------------------------------------------------------
template <int AF32, int ACT, int OUTBF16, int STORET>
__global__ __launch_bounds__(256) void gemm_wmma(
    const void* __restrict__ Aptr, const __bf16* __restrict__ Bw,
    const float* __restrict__ bias, void* __restrict__ Cptr,
    int M, int N, int K, int lda, int ldb,
    long batchA, long batchB, long batchC, int Sdim) {
  __shared__ __bf16 bstage[2][256 * RSTR];
  const int tid = threadIdx.x;
  const int l   = tid & 31;
  const int w   = tid >> 5;
  const int mbase   = blockIdx.y * 64 + (w >> 2) * 32;
  const int nlocal  = (w & 3) * 64;
  const int nblock0 = blockIdx.x * 256;
  const int z = blockIdx.z;
  const __bf16* Ab = (const __bf16*)Aptr + z * batchA;
  const float*  Af = (const float*)Aptr + z * batchA;
  const __bf16* Bp = Bw + z * batchB;
  float*  Cf = (float*)Cptr + z * batchC;
  __bf16* Cb = (__bf16*)Cptr + z * batchC;

  const int lh = l & 15, hh = l >> 4;
  int mA[2];
#pragma unroll
  for (int i = 0; i < 2; ++i) {
    mA[i] = mbase + i * 16 + lh;
    if (mA[i] > M - 1) mA[i] = M - 1;
  }

  v8f acc[2][4];
#pragma unroll
  for (int i = 0; i < 2; ++i)
#pragma unroll
    for (int t = 0; t < 4; ++t) acc[i][t] = vzero8();

  const int nk = K / 32;

  auto compute_step = [&](const __bf16* bs, int k0) {
    v16bf fa[2], fb[4];
#pragma unroll
    for (int i = 0; i < 2; ++i) {
      if (AF32) {
        const float* p = Af + (long)mA[i] * lda + k0 + hh * 8;
#pragma unroll
        for (int e = 0; e < 8; ++e) { fa[i][e] = f2bf(p[e]); fa[i][8 + e] = f2bf(p[16 + e]); }
        __builtin_prefetch(p + 32, 0, 3);
      } else {
        const __bf16* p = Ab + (long)mA[i] * lda + k0 + hh * 8;
#pragma unroll
        for (int e = 0; e < 8; ++e) { fa[i][e] = p[e]; fa[i][8 + e] = p[16 + e]; }
        __builtin_prefetch(p + 32, 0, 3);
      }
    }
#pragma unroll
    for (int t = 0; t < 4; ++t) {
      const __bf16* q = bs + (nlocal + t * 16 + lh) * RSTR + hh * 8;
#pragma unroll
      for (int e = 0; e < 8; ++e) { fb[t][e] = q[e]; fb[t][8 + e] = q[16 + e]; }
    }
#pragma unroll
    for (int i = 0; i < 2; ++i)
#pragma unroll
      for (int t = 0; t < 4; ++t)
        acc[i][t] = __builtin_amdgcn_wmma_f32_16x16x32_bf16(false, fa[i], false, fb[t],
                                                            (short)0, acc[i][t], false, false);
  };

#if TICD_TDM
  if (w == 0)
    tdm_load_b_tile(Bp + (long)nblock0 * ldb, (unsigned)(uintptr_t)&bstage[0][0],
                    N - nblock0, ldb);
  for (int kk = 0; kk < nk; ++kk) {
    if (w == 0) __builtin_amdgcn_s_wait_tensorcnt((short)0);
    __syncthreads();
    if (w == 0 && kk + 1 < nk)
      tdm_load_b_tile(Bp + (long)nblock0 * ldb + (kk + 1) * 32,
                      (unsigned)(uintptr_t)&bstage[(kk + 1) & 1][0], N - nblock0, ldb);
    compute_step(bstage[kk & 1], kk * 32);
  }
#else
  for (int kk = 0; kk < nk; ++kk) {
    __syncthreads();
    {  // cooperative stage: thread tid copies row tid of the B tile
      const int n = nblock0 + tid;
      __bf16* dst = &bstage[0][tid * RSTR];
      if (n < N) {
        const __bf16* src = Bp + (long)n * ldb + kk * 32;
#pragma unroll
        for (int e = 0; e < 32; ++e) dst[e] = src[e];
      } else {
        const __bf16 zz = f2bf(0.f);
#pragma unroll
        for (int e = 0; e < 32; ++e) dst[e] = zz;
      }
    }
    __syncthreads();
    compute_step(bstage[0], kk * 32);
  }
#endif

  // ---- store: C layout lane=col n, row = 8*hh + r ----
#pragma unroll
  for (int i = 0; i < 2; ++i) {
#pragma unroll
    for (int t = 0; t < 4; ++t) {
      const int col = nblock0 + nlocal + t * 16 + lh;
      float bb = 0.f;
      if (bias != nullptr && col < N) bb = bias[col];
#pragma unroll
      for (int r = 0; r < 8; ++r) {
        const int row = mbase + i * 16 + hh * 8 + r;
        if (row < M && col < N) {
          float vv = acc[i][t][r] + bb;
          if (ACT == 1) vv = fmaxf(vv, 0.f);
          if (ACT == 2) vv = tanhf(vv);
          if (STORET) {
            const long bb2 = row / Sdim;
            const long s   = row - bb2 * Sdim;
            Cf[(bb2 * (long)N + col) * (long)Sdim + s] = vv;
          } else if (OUTBF16) {
            Cb[(long)row * N + col] = f2bf(vv);
          } else {
            Cf[(long)row * N + col] = vv;
          }
        }
      }
    }
  }
}

// ------------------------------------------------------------------
// Flash attention: 1 wave per 16-row q tile, online softmax over 32-col chunks.
// qkv: [B,S,3E] bf16; vT: [B*H*DH, S] bf16 (pre-transposed V).
// ------------------------------------------------------------------
__global__ __launch_bounds__(128) void attn_kernel(const __bf16* __restrict__ qkv,
                                                   const __bf16* __restrict__ vT,
                                                   __bf16* __restrict__ obuf) {
  __shared__ __bf16 pstage[4][16 * 32];
  const int wave = threadIdx.x >> 5;
  const int wid  = blockIdx.x * 4 + wave;
  const int l  = threadIdx.x & 31;
  const int lh = l & 15, hh = l >> 4;
  const int qt = wid & 63;
  const int h  = (wid >> 6) & 7;
  const int b  = wid >> 9;
  __bf16* ps = pstage[wave];

  const __bf16* Q  = qkv + (long)b * S * LD3E + h * DH;
  const __bf16* Kp = Q + E;
  const __bf16* Vt = vT + ((long)b * H + h) * DH * S;

  // Q A-fragments (16 x 64 -> two 16x32 fragments)
  const int qrow = qt * 16 + lh;
  v16bf aq0, aq1;
  {
    const __bf16* qp = Q + (long)qrow * LD3E;
#pragma unroll
    for (int e = 0; e < 8; ++e) {
      aq0[e] = qp[hh * 16 + e];       aq0[8 + e] = qp[hh * 16 + 8 + e];
      aq1[e] = qp[32 + hh * 16 + e];  aq1[8 + e] = qp[40 + hh * 16 + e];
    }
  }

  float mrow[8], lsum[8];
#pragma unroll
  for (int r = 0; r < 8; ++r) { mrow[r] = -1e30f; lsum[r] = 0.f; }
  v8f o[4] = {vzero8(), vzero8(), vzero8(), vzero8()};

  for (int kt = 0; kt < 32; ++kt) {           // 32 chunks of 32 key columns
    v8f s0 = vzero8(), s1 = vzero8();
    {
      const __bf16* p0 = Kp + (long)(kt * 32 + lh) * LD3E + hh * 16;
      const __bf16* p1 = Kp + (long)(kt * 32 + 16 + lh) * LD3E + hh * 16;
      v16bf bk00, bk01, bk10, bk11;
#pragma unroll
      for (int e = 0; e < 16; ++e) {
        bk00[e] = p0[e]; bk01[e] = p0[32 + e];
        bk10[e] = p1[e]; bk11[e] = p1[32 + e];
      }
      s0 = __builtin_amdgcn_wmma_f32_16x16x32_bf16(false, aq0, false, bk00, (short)0, s0, false, false);
      s0 = __builtin_amdgcn_wmma_f32_16x16x32_bf16(false, aq1, false, bk01, (short)0, s0, false, false);
      s1 = __builtin_amdgcn_wmma_f32_16x16x32_bf16(false, aq0, false, bk10, (short)0, s1, false, false);
      s1 = __builtin_amdgcn_wmma_f32_16x16x32_bf16(false, aq1, false, bk11, (short)0, s1, false, false);
    }
    // online softmax per row (tile rows live across one 16-lane half)
    float pr0[8], pr1[8];
#pragma unroll
    for (int r = 0; r < 8; ++r) {
      float v0 = s0[r] * 0.125f, v1 = s1[r] * 0.125f;   // 1/sqrt(DH)
      float mx = fmaxf(v0, v1);
#pragma unroll
      for (int d = 1; d < 16; d <<= 1) mx = fmaxf(mx, __shfl_xor(mx, d, 16));
      float newm = fmaxf(mrow[r], mx);
      float corr = __expf(mrow[r] - newm);
      mrow[r] = newm;
      float e0 = __expf(v0 - newm), e1 = __expf(v1 - newm);
      float psum = e0 + e1;
#pragma unroll
      for (int d = 1; d < 16; d <<= 1) psum += __shfl_xor(psum, d, 16);
      lsum[r] = lsum[r] * corr + psum;
      o[0][r] *= corr; o[1][r] *= corr; o[2][r] *= corr; o[3][r] *= corr;
      pr0[r] = e0; pr1[r] = e1;
    }
    // transpose P (16x32) through LDS into an A-fragment
#pragma unroll
    for (int r = 0; r < 8; ++r) {
      ps[(hh * 8 + r) * 32 + lh]      = f2bf(pr0[r]);
      ps[(hh * 8 + r) * 32 + 16 + lh] = f2bf(pr1[r]);
    }
    __syncthreads();
    v16bf ap;
    {
      const __bf16* pp = ps + lh * 32 + hh * 16;
#pragma unroll
      for (int e = 0; e < 8; ++e) { ap[e] = pp[e]; ap[8 + e] = pp[8 + e]; }
    }
    // O += P (16x32) * V (32x64): B-frags contiguous from vT
#pragma unroll
    for (int j = 0; j < 4; ++j) {
      v16bf bv;
      const __bf16* vp = Vt + (long)(j * 16 + lh) * S + kt * 32 + hh * 16;
#pragma unroll
      for (int e = 0; e < 16; ++e) bv[e] = vp[e];
      o[j] = __builtin_amdgcn_wmma_f32_16x16x32_bf16(false, ap, false, bv, (short)0, o[j], false, false);
    }
    __syncthreads();
  }

  // finalize & store bf16
#pragma unroll
  for (int r = 0; r < 8; ++r) {
    float inv = 1.f / lsum[r];
    int srow = qt * 16 + hh * 8 + r;
    long base = ((long)b * S + srow) * (long)E + h * DH;
#pragma unroll
    for (int j = 0; j < 4; ++j) obuf[base + j * 16 + lh] = f2bf(o[j][r] * inv);
  }
}

// ------------------------------------------------------------------
// residual + LayerNorm: x = LN(x + t)*g + b  -> writes f32 x and bf16 xb
// ------------------------------------------------------------------
__global__ __launch_bounds__(32) void ln_kernel(const float* __restrict__ xin,
                                                const float* __restrict__ t,
                                                const float* __restrict__ g,
                                                const float* __restrict__ bta,
                                                float* __restrict__ xo,
                                                __bf16* __restrict__ xob) {
  const long row = blockIdx.x;
  const int l = threadIdx.x;
  const float* xr = xin + row * E;
  const float* tr = t + row * E;
  float v[16];
  float s = 0.f;
#pragma unroll
  for (int i = 0; i < 16; ++i) { v[i] = xr[i * 32 + l] + tr[i * 32 + l]; s += v[i]; }
#pragma unroll
  for (int d = 1; d < 32; d <<= 1) s += __shfl_xor(s, d, 32);
  const float mu = s * (1.f / E);
  float vs = 0.f;
#pragma unroll
  for (int i = 0; i < 16; ++i) { float dd = v[i] - mu; vs += dd * dd; }
#pragma unroll
  for (int d = 1; d < 32; d <<= 1) vs += __shfl_xor(vs, d, 32);
  const float rinv = rsqrtf(vs * (1.f / E) + 1e-5f);
#pragma unroll
  for (int i = 0; i < 16; ++i) {
    int c = i * 32 + l;
    float y = (v[i] - mu) * rinv * g[c] + bta[c];
    xo[row * E + c] = y;
    xob[row * E + c] = f2bf(y);
  }
}

// ------------------------------------------------------------------
// label softmax over seq dim, in place on d_out attn region [B,C,S]
// ------------------------------------------------------------------
__global__ __launch_bounds__(256) void smax_kernel(float* __restrict__ attn) {
  const long rid = (long)blockIdx.x * 8 + (threadIdx.x >> 5);
  if (rid >= (long)B * Cc) return;
  const int l = threadIdx.x & 31;
  float* p = attn + rid * S;
  float v[32];
  float mx = -1e30f;
#pragma unroll
  for (int i = 0; i < 32; ++i) { v[i] = p[i * 32 + l]; mx = fmaxf(mx, v[i]); }
#pragma unroll
  for (int d = 1; d < 32; d <<= 1) mx = fmaxf(mx, __shfl_xor(mx, d, 32));
  float s = 0.f;
#pragma unroll
  for (int i = 0; i < 32; ++i) { v[i] = __expf(v[i] - mx); s += v[i]; }
#pragma unroll
  for (int d = 1; d < 32; d <<= 1) s += __shfl_xor(s, d, 32);
  const float inv = 1.f / s;
#pragma unroll
  for (int i = 0; i < 32; ++i) p[i * 32 + l] = v[i] * inv;
}

// ------------------------------------------------------------------
// outputs[b,c] = weighted[b,c,:] . fc_w[c,:] + fc_b[c]
// ------------------------------------------------------------------
__global__ __launch_bounds__(256) void dot_kernel(const float* __restrict__ wtd,
                                                  const float* __restrict__ fcw,
                                                  const float* __restrict__ fcb,
                                                  float* __restrict__ out) {
  const long rid = (long)blockIdx.x * 8 + (threadIdx.x >> 5);
  if (rid >= (long)B * Cc) return;
  const int l = threadIdx.x & 31;
  const int c = (int)(rid % Cc);
  const float* a = wtd + rid * E;
  const float* w = fcw + (long)c * E;
  float s = 0.f;
#pragma unroll
  for (int i = 0; i < 16; ++i) s += a[i * 32 + l] * w[i * 32 + l];
#pragma unroll
  for (int d = 1; d < 32; d <<= 1) s += __shfl_xor(s, d, 32);
  if (l == 0) out[rid] = s + fcb[c];
}

}  // namespace ticd

// ==================================================================
extern "C" void kernel_launch(void* const* d_in, const int* in_sizes, int n_in,
                              void* d_out, int out_size, void* d_ws, size_t ws_size,
                              hipStream_t stream) {
  using namespace ticd;
  (void)in_sizes; (void)n_in; (void)out_size; (void)ws_size;

  const float* inputs     = (const float*)d_in[0];
  const float* pe         = (const float*)d_in[1];
  const float* label_mat  = (const float*)d_in[2];
  const float* l1_w       = (const float*)d_in[3];
  const float* l1_b       = (const float*)d_in[4];
  const float* fc_w       = (const float*)d_in[5];
  const float* fc_b       = (const float*)d_in[6];
  const float* in_proj_w  = (const float*)d_in[7];
  const float* in_proj_b  = (const float*)d_in[8];
  const float* out_proj_w = (const float*)d_in[9];
  const float* out_proj_b = (const float*)d_in[10];
  const float* ln1_w      = (const float*)d_in[11];
  const float* ln1_b      = (const float*)d_in[12];
  const float* ln2_w      = (const float*)d_in[13];
  const float* ln2_b      = (const float*)d_in[14];
  const float* lin1_w     = (const float*)d_in[15];
  const float* lin1_b     = (const float*)d_in[16];
  const float* lin2_w     = (const float*)d_in[17];
  const float* lin2_b     = (const float*)d_in[18];

  float* out   = (float*)d_out;                    // [B*C] logits
  float* attnw = out + (size_t)B * Cc;             // [B,C,S] attention weights

  // ---- workspace layout ----
  char* wsp = (char*)d_ws;
  auto wsa = [&](size_t bytes) -> char* {
    char* p = wsp;
    wsp += (bytes + 255) & ~size_t(255);
    return p;
  };
  __bf16* w_qkv = (__bf16*)wsa((size_t)L * 3 * E * E * 2);
  __bf16* w_o   = (__bf16*)wsa((size_t)L * E * E * 2);
  __bf16* w_1   = (__bf16*)wsa((size_t)L * DFF * E * 2);
  __bf16* w_2   = (__bf16*)wsa((size_t)L * E * DFF * 2);
  __bf16* w_l1  = (__bf16*)wsa((size_t)E * E * 2);
  __bf16* w_lab = (__bf16*)wsa((size_t)Cc * E * 2);
  float*  x     = (float*)wsa((size_t)B * S * E * 4);
  __bf16* xb    = (__bf16*)wsa((size_t)B * S * E * 2);
  __bf16* qkv   = (__bf16*)wsa((size_t)B * S * 3 * E * 2);
  __bf16* vT    = (__bf16*)wsa((size_t)B * E * S * 2);
  __bf16* obuf  = (__bf16*)wsa((size_t)B * S * E * 2);
  float*  tmp   = (float*)wsa((size_t)B * S * E * 4);
  __bf16* ff1   = (__bf16*)wsa((size_t)B * S * DFF * 2);
  __bf16* o1b   = (__bf16*)wsa((size_t)B * S * E * 2);
  __bf16* xbT   = (__bf16*)wsa((size_t)B * E * S * 2);
  float*  wtd   = (float*)wsa((size_t)B * Cc * E * 4);

  // ---- weight converts (f32 -> bf16) ----
  auto cvt = [&](const float* src, __bf16* dst, long n) {
    int blocks = (int)((n + 255) / 256);
    if (blocks > 4096) blocks = 4096;
    cvt_kernel<<<blocks, 256, 0, stream>>>(src, dst, n);
  };
  cvt(in_proj_w,  w_qkv, (long)L * 3 * E * E);
  cvt(out_proj_w, w_o,   (long)L * E * E);
  cvt(lin1_w,     w_1,   (long)L * DFF * E);
  cvt(lin2_w,     w_2,   (long)L * E * DFF);
  cvt(l1_w,       w_l1,  (long)E * E);
  cvt(label_mat,  w_lab, (long)Cc * E);

  // ---- embed ----
  {
    long n = (long)B * S * E;
    embed_kernel<<<(int)((n + 255) / 256), 256, 0, stream>>>(inputs, pe, x, xb);
  }

  const int M = B * S;          // 8192
  const long nTE = (long)B * E * S;
  // ---- encoder layers ----
  for (int i = 0; i < L; ++i) {
    // QKV = xb @ Wqkv^T + b      [8192 x 1536] bf16
    gemm_wmma<0, 0, 1, 0><<<dim3((3 * E + 255) / 256, (M + 63) / 64, 1), 256, 0, stream>>>(
        xb, w_qkv + (size_t)i * 3 * E * E, in_proj_b + i * 3 * E, qkv,
        M, 3 * E, E, E, E, 0, 0, 0, 0);
    // transpose V for contiguous P@V operand loads
    transpose_v_kernel<<<(int)((nTE + 255) / 256), 256, 0, stream>>>(qkv, vT);
    // attention -> obuf
    attn_kernel<<<(B * H * (S / 16)) / 4, 128, 0, stream>>>(qkv, vT, obuf);
    // out proj -> tmp (f32)
    gemm_wmma<0, 0, 0, 0><<<dim3((E + 255) / 256, (M + 63) / 64, 1), 256, 0, stream>>>(
        obuf, w_o + (size_t)i * E * E, out_proj_b + i * E, tmp,
        M, E, E, E, E, 0, 0, 0, 0);
    // x = LN1(x + tmp)
    ln_kernel<<<M, 32, 0, stream>>>(x, tmp, ln1_w + i * E, ln1_b + i * E, x, xb);
    // ff1 = relu(xb @ W1^T + b1)  [8192 x 2048] bf16
    gemm_wmma<0, 1, 1, 0><<<dim3((DFF + 255) / 256, (M + 63) / 64, 1), 256, 0, stream>>>(
        xb, w_1 + (size_t)i * DFF * E, lin1_b + i * DFF, ff1,
        M, DFF, E, E, E, 0, 0, 0, 0);
    // tmp = ff1 @ W2^T + b2       [8192 x 512] f32
    gemm_wmma<0, 0, 0, 0><<<dim3((E + 255) / 256, (M + 63) / 64, 1), 256, 0, stream>>>(
        ff1, w_2 + (size_t)i * E * DFF, lin2_b + i * E, tmp,
        M, E, DFF, DFF, DFF, 0, 0, 0, 0);
    // x = LN2(x + tmp)
    ln_kernel<<<M, 32, 0, stream>>>(x, tmp, ln2_w + i * E, ln2_b + i * E, x, xb);
  }

  // ---- label attention ----
  // o1 = tanh(xb @ l1_w^T + l1_b)  bf16
  gemm_wmma<0, 2, 1, 0><<<dim3((E + 255) / 256, (M + 63) / 64, 1), 256, 0, stream>>>(
      xb, w_l1, l1_b, o1b, M, E, E, E, E, 0, 0, 0, 0);
  // scores = o1 @ label_mat^T, stored transposed into d_out as [B,C,S]
  gemm_wmma<0, 0, 0, 1><<<dim3((Cc + 255) / 256, (M + 63) / 64, 1), 256, 0, stream>>>(
      o1b, w_lab, nullptr, attnw, M, Cc, E, E, E, 0, 0, 0, S);
  // softmax over seq dim, in place in d_out
  smax_kernel<<<(B * Cc) / 8, 256, 0, stream>>>(attnw);
  // xbT = transpose(xb) per batch -> weight-layout operand for attn@x
  transpose_x_kernel<<<(int)((nTE + 255) / 256), 256, 0, stream>>>(xb, xbT);
  // weighted[b] = attn[b] (C x S, f32) @ xbT[b]^T (S x E)  -> wtd [B,C,E] f32
  gemm_wmma<1, 0, 0, 0><<<dim3((E + 255) / 256, (Cc + 63) / 64, B), 256, 0, stream>>>(
      attnw, xbT, nullptr, wtd, Cc, E, S, S, S,
      (long)Cc * S, (long)E * S, (long)Cc * E, 0);
  // outputs[b,c] = weighted[b,c,:] . fc_w[c,:] + fc_b[c]
  dot_kernel<<<(B * Cc) / 8, 256, 0, stream>>>(wtd, fc_w, fc_b, out);
}
